// MDLSTMConvBlock_57217554317585
// MI455X (gfx1250) — matrix operand order repaired and checked by reading, not compiled, not run on hardware
//
#include <hip/hip_runtime.h>
#include <hip/hip_bf16.h>

#define BD   32      // batch
#define CIN  32
#define HH   48
#define WW   160
#define CL   32
#define Z5   160     // 5*CL
#define CO   64
#define OH   24
#define OW   80
#define EPSN 1e-5f

typedef __attribute__((ext_vector_type(16))) _Float16 v16h;
typedef __attribute__((ext_vector_type(8)))  _Float16 v8h;
typedef __attribute__((ext_vector_type(8)))  float    v8f;

// ---------------------------------------------------------------------------
// V_WMMA_F32_16X16X32_F16 fragment maps (wave32) as used in this file:
//  A (16x32 f16): lane -> row m=lane&15, half=lane>>4.
//    elem e : K = half*16 + e  (contiguous, matches B packing)
//    => from a row-major f16 row: one contiguous 32B chunk at half*16
//  B (32x16 f16): col n=lane&15; elem e -> K = e + half*16 (prepacked per lane)
//  C/D (16x16 f32): VGPR r -> row = r + 8*(lane>>4), col = lane&15
// ---------------------------------------------------------------------------
__device__ __forceinline__ v16h load_a_frag_f16(const _Float16* __restrict__ row,
                                                int half) {
    return *(const v16h*)(row + half * 16);
}

__device__ __forceinline__ float sigmoidf_(float x) {
    return 1.0f / (1.0f + __expf(-x));
}

// ---------------------------------------------------------------------------
// Prep: pack Wx/Ul/Ut into fragment-ordered f16.
// wpack[mat][d][nt][lane][e],  mat in {0:Wx, 1:Ul, 2:Ut}
// value = M[d][k][nt*16+n], k = e + (lane>>4)*16, n = lane&15
// ---------------------------------------------------------------------------
__global__ void pack_weights_kernel(const float* __restrict__ Wx,
                                    const float* __restrict__ Ul,
                                    const float* __restrict__ Ut,
                                    _Float16* __restrict__ wpack)
{
    const int t = blockIdx.x * blockDim.x + threadIdx.x;
    if (t >= 3 * 4 * 10 * 32 * 16) return;
    const int e    = t & 15;
    const int lane = (t >> 4) & 31;
    const int nt   = (t >> 9) % 10;
    const int d    = (t / (16 * 32 * 10)) % 4;
    const int mat  = t / (16 * 32 * 10 * 4);
    const int n = lane & 15, half = lane >> 4;
    const int k = e + half * 16;
    const float* M = (mat == 0 ? Wx : (mat == 1 ? Ul : Ut)) + (size_t)d * 32 * Z5;
    wpack[t] = (_Float16)M[(size_t)k * Z5 + nt * 16 + n];
}

// cpack[nt][kh][kw][lane][e] = conv_w[co=nt*16+n][ci=e+half*16][kh][kw]
__global__ void pack_convw_kernel(const float* __restrict__ conv_w,
                                  _Float16* __restrict__ cpack)
{
    const int t = blockIdx.x * blockDim.x + threadIdx.x;
    if (t >= 4 * 2 * 2 * 32 * 16) return;
    const int e    = t & 15;
    const int lane = (t >> 4) & 31;
    const int kw   = (t >> 9) & 1;
    const int kh   = (t >> 10) & 1;
    const int nt   = (t >> 11) & 3;
    const int n = lane & 15, half = lane >> 4;
    const int k = e + half * 16;
    cpack[t] = (_Float16)conv_w[(((size_t)(nt * 16 + n) * CL + k) * 2 + kh) * 2 + kw];
}

// xT[h][w][b][k] (f16) from x[b][k][h][w] (f32)
__global__ void xpack_kernel(const float* __restrict__ x, _Float16* __restrict__ xT)
{
    const size_t total = (size_t)HH * WW * BD * CIN;
    size_t i = (size_t)blockIdx.x * blockDim.x + threadIdx.x;
    if (i >= total) return;
    const int k = i & 31;
    size_t t = i >> 5;
    const int b = t & 31; t >>= 5;
    const int w = (int)(t % WW);
    const int h = (int)(t / WW);
    xT[i] = (_Float16)x[(((size_t)b * CIN + k) * HH + h) * WW + w];
}

__global__ void init_progress_kernel(int* __restrict__ progress)
{
    const int i = blockIdx.x * blockDim.x + threadIdx.x;
    if (i < 4 * HH) progress[i] = 0;
}

// ---------------------------------------------------------------------------
// Persistent MDLSTM: one workgroup per (direction, row). 10 waves (320 thr).
// Left dependency lives in LDS; top dependency synced through a per-row
// progress counter (release store / acquire load + s_sleep spin).
// Per column, wave nt does 2 M-tiles x 3 WMMAs (Wx, Ul, Ut), all f16 frags.
// ---------------------------------------------------------------------------
__global__ void __launch_bounds__(320)
mdlstm_row_kernel(const _Float16* __restrict__ xT,
                  const _Float16* __restrict__ wpack,
                  const float* __restrict__ bias,
                  float* __restrict__ hst,
                  float* __restrict__ cst,
                  _Float16* __restrict__ hst16,
                  int* __restrict__ progress)
{
    __shared__ __align__(16) float    zld[BD * Z5];    // 20 KB
    __shared__ __align__(16) _Float16 hprev[BD * CL];  // 2 KB (h at w-1)
    __shared__ __align__(16) float    cprev[BD * CL];  // 4 KB (c at w-1)

    const int r = blockIdx.x;         // row in scan coords
    const int d = blockIdx.y;         // direction
    const int lane = threadIdx.x & 31;
    const int nt   = threadIdx.x >> 5;   // 0..9: gate-column tile
    const int n    = lane & 15;
    const int half = lane >> 4;

    for (int i = threadIdx.x; i < BD * CL; i += 320) {
        hprev[i] = (_Float16)0.0f;
        cprev[i] = 0.0f;
    }
    __syncthreads();

    // loop-invariant fragments / scalars
    const v16h bWx = *(const v16h*)(wpack + ((((size_t)0 * 4 + d) * 10 + nt) * 32 + lane) * 16);
    const v16h bUl = *(const v16h*)(wpack + ((((size_t)1 * 4 + d) * 10 + nt) * 32 + lane) * 16);
    const v16h bUt = *(const v16h*)(wpack + ((((size_t)2 * 4 + d) * 10 + nt) * 32 + lane) * 16);
    const float bv = bias[(size_t)d * Z5 + nt * 16 + n];

    const size_t cellstride = (size_t)BD * CL;
    const int hp = (d >= 2) ? (HH - 1 - r) : r;

    int* myprog = progress + d * HH + r;
    int* topprog = progress + d * HH + (r - 1);

    for (int w = 0; w < WW; ++w) {
        if (r > 0) {                    // uniform branch per workgroup
            if (threadIdx.x == 0) {
                while (__hip_atomic_load(topprog, __ATOMIC_ACQUIRE,
                                         __HIP_MEMORY_SCOPE_AGENT) <= w) {
                    __builtin_amdgcn_s_sleep(1);
                }
            }
            __syncthreads();
        }

        const int wp = (d & 1) ? (WW - 1 - w) : w;
        const size_t base  = (((size_t)d * HH + r) * WW + w) * cellstride;
        const size_t baseT = base - (size_t)WW * cellstride;   // (r-1, w)
        const _Float16* xcell = xT + (((size_t)hp * WW + wp) * BD) * CIN;
        const _Float16* htop  = hst16 + baseT;

#pragma unroll
        for (int mt = 0; mt < 2; ++mt) {
            const int mrow = mt * 16 + (lane & 15);
            v8f acc = { bv, bv, bv, bv, bv, bv, bv, bv };

            // x @ Wx
            {
                v16h a = load_a_frag_f16(xcell + (size_t)mrow * CIN, half);
                acc = __builtin_amdgcn_wmma_f32_16x16x32_f16(
                          false, a, false, bWx, (short)0, acc, false, false);
            }
            // h_left @ Ul  (LDS; zeros at w==0 -> unconditional, EXEC full)
            {
                v16h a = load_a_frag_f16(hprev + (size_t)mrow * CL, half);
                acc = __builtin_amdgcn_wmma_f32_16x16x32_f16(
                          false, a, false, bUl, (short)0, acc, false, false);
            }
            // h_top @ Ut
            if (r > 0) {
                v16h a = load_a_frag_f16(htop + (size_t)mrow * CL, half);
                acc = __builtin_amdgcn_wmma_f32_16x16x32_f16(
                          false, a, false, bUt, (short)0, acc, false, false);
            }

#pragma unroll
            for (int rr = 0; rr < 8; ++rr) {
                zld[(mt * 16 + rr + 8 * half) * Z5 + nt * 16 + n] = acc[rr];
            }
        }
        __syncthreads();

        // gates: [i, f_left, f_top, o, g] chunks of 32 along the 160 dim
        for (int idx = threadIdx.x; idx < BD * CL; idx += 320) {
            const int b  = idx >> 5;
            const int ch = idx & 31;
            const float zi  = zld[b * Z5 + 0 * CL + ch];
            const float zfl = zld[b * Z5 + 1 * CL + ch];
            const float zft = zld[b * Z5 + 2 * CL + ch];
            const float zo  = zld[b * Z5 + 3 * CL + ch];
            const float zg  = zld[b * Z5 + 4 * CL + ch];
            const float cl_ = cprev[idx];
            const float ct_ = (r > 0) ? cst[baseT + idx] : 0.0f;
            const float c = sigmoidf_(zfl) * cl_ + sigmoidf_(zft) * ct_
                          + sigmoidf_(zi)  * tanhf(zg);
            const float h = sigmoidf_(zo) * tanhf(c);
            hst[base + idx]   = h;
            cst[base + idx]   = c;
            hst16[base + idx] = (_Float16)h;
            hprev[idx] = (_Float16)h;
            cprev[idx] = c;
        }
        __threadfence();     // make h/c column visible device-wide
        __syncthreads();
        if (threadIdx.x == 0) {
            __hip_atomic_store(myprog, w + 1, __ATOMIC_RELEASE,
                               __HIP_MEMORY_SCOPE_AGENT);
        }
    }
}

// ---------------------------------------------------------------------------
// Sum 4 direction outputs back to natural coords, store f16 [h][w][b][c]
// ---------------------------------------------------------------------------
__global__ void hsum_kernel(const float* __restrict__ hst, _Float16* __restrict__ hsum16)
{
    const size_t total = (size_t)HH * WW * BD * CL;
    size_t i = (size_t)blockIdx.x * blockDim.x + threadIdx.x;
    if (i >= total) return;
    const int c = i & 31;
    size_t t = i >> 5;
    const int b = t & 31; t >>= 5;
    const int w = (int)(t % WW);
    const int h = (int)(t / WW);
    float s = 0.0f;
#pragma unroll
    for (int d = 0; d < 4; ++d) {
        const int hp = (d >= 2) ? (HH - 1 - h) : h;
        const int wp = (d & 1)  ? (WW - 1 - w) : w;
        s += hst[((((size_t)d * HH + hp) * WW + wp) * BD + b) * CL + c];
    }
    hsum16[i] = (_Float16)s;
}

// ---------------------------------------------------------------------------
// Conv 2x2/s2 as im2col WMMA GEMM: M=B*OH*OW, K=128 (4x32), N=64.
// grid = (3840, 4), block = 32. Fused +4*conv_b and tanh at the store.
// ---------------------------------------------------------------------------
__global__ void __launch_bounds__(32)
conv_wmma_kernel(const _Float16* __restrict__ hsum16,
                 const _Float16* __restrict__ cpack,
                 const float* __restrict__ conv_b,
                 float* __restrict__ y)
{
    const int mt = blockIdx.x;
    const int nt = blockIdx.y;
    const int lane = threadIdx.x & 31;
    const int n = lane & 15, half = lane >> 4;

    v8f acc = { 0.f, 0.f, 0.f, 0.f, 0.f, 0.f, 0.f, 0.f };

    const int m   = mt * 16 + (lane & 15);
    const int b   = m / (OH * OW);
    const int rem = m % (OH * OW);
    const int oh  = rem / OW;
    const int ow  = rem % OW;

#pragma unroll
    for (int kh = 0; kh < 2; ++kh) {
#pragma unroll
        for (int kw = 0; kw < 2; ++kw) {
            const int h = oh * 2 + kh;
            const int w = ow * 2 + kw;
            v16h a = load_a_frag_f16(
                hsum16 + (((size_t)h * WW + w) * BD + b) * CL, half);
            v16h bf = *(const v16h*)(cpack +
                ((((size_t)nt * 2 + kh) * 2 + kw) * 32 + lane) * 16);
            acc = __builtin_amdgcn_wmma_f32_16x16x32_f16(
                      false, a, false, bf, (short)0, acc, false, false);
        }
    }

    const int co = nt * 16 + n;
    const float bb = 4.0f * conv_b[co];
#pragma unroll
    for (int rr = 0; rr < 8; ++rr) {
        const int mm   = mt * 16 + rr + 8 * half;
        const int b2   = mm / (OH * OW);
        const int rem2 = mm % (OH * OW);
        y[((size_t)b2 * CO + co) * (OH * OW) + rem2] = tanhf(acc[rr] + bb);
    }
}

// ---------------------------------------------------------------------------
// Per-channel mean / rstd over (B, OH, OW). grid = 64, block = 256.
// ---------------------------------------------------------------------------
__global__ void __launch_bounds__(256)
stats_kernel(const float* __restrict__ y, float* __restrict__ stats)
{
    const int co = blockIdx.x;
    const int Nper = BD * OH * OW;
    float s = 0.f, s2 = 0.f;
    for (int i = threadIdx.x; i < Nper; i += 256) {
        const int b   = i / (OH * OW);
        const int rem = i % (OH * OW);
        const float v = y[((size_t)b * CO + co) * (OH * OW) + rem];
        s += v; s2 += v * v;
    }
    __shared__ float sh[256], sh2[256];
    sh[threadIdx.x] = s; sh2[threadIdx.x] = s2;
    __syncthreads();
    for (int off = 128; off > 0; off >>= 1) {
        if (threadIdx.x < off) {
            sh[threadIdx.x]  += sh[threadIdx.x + off];
            sh2[threadIdx.x] += sh2[threadIdx.x + off];
        }
        __syncthreads();
    }
    if (threadIdx.x == 0) {
        const float mean = sh[0] / (float)Nper;
        const float var  = sh2[0] / (float)Nper - mean * mean;
        stats[co]      = mean;
        stats[64 + co] = rsqrtf(var + EPSN);
    }
}

__global__ void norm_kernel(const float* __restrict__ y,
                            const float* __restrict__ stats,
                            const float* __restrict__ gamma,
                            const float* __restrict__ beta,
                            float* __restrict__ out)
{
    const size_t total = (size_t)BD * CO * OH * OW;
    size_t i = (size_t)blockIdx.x * blockDim.x + threadIdx.x;
    if (i >= total) return;
    const int co = (int)((i / (OH * OW)) % CO);
    out[i] = gamma[co] * (y[i] - stats[co]) * stats[64 + co] + beta[co];
}

// ---------------------------------------------------------------------------
extern "C" void kernel_launch(void* const* d_in, const int* in_sizes, int n_in,
                              void* d_out, int out_size, void* d_ws, size_t ws_size,
                              hipStream_t stream)
{
    const float* x      = (const float*)d_in[0];
    const float* Wx     = (const float*)d_in[1];
    const float* Ul     = (const float*)d_in[2];
    const float* Ut     = (const float*)d_in[3];
    const float* bias   = (const float*)d_in[4];
    const float* conv_w = (const float*)d_in[5];
    const float* conv_b = (const float*)d_in[6];
    const float* gamma  = (const float*)d_in[7];
    const float* beta   = (const float*)d_in[8];
    float* out = (float*)d_out;

    const size_t cellTotal = (size_t)HH * WW * BD * CL;   // 7,864,320
    char* p = (char*)d_ws;
    auto alloc = [&](size_t bytes) -> char* {
        char* q = p;
        p += (bytes + 255) & ~(size_t)255;
        return q;
    };
    float*    hst      = (float*)   alloc(4 * cellTotal * sizeof(float));
    float*    cst      = (float*)   alloc(4 * cellTotal * sizeof(float));
    _Float16* hst16    = (_Float16*)alloc(4 * cellTotal * sizeof(_Float16));
    _Float16* xT       = (_Float16*)alloc(cellTotal * sizeof(_Float16));
    _Float16* hsum16   = (_Float16*)alloc(cellTotal * sizeof(_Float16));
    float*    y        = (float*)   alloc((size_t)BD * CO * OH * OW * sizeof(float));
    float*    stats    = (float*)   alloc(128 * sizeof(float));
    _Float16* wpack    = (_Float16*)alloc(3 * 4 * 10 * 32 * 16 * sizeof(_Float16));
    _Float16* cpack    = (_Float16*)alloc(4 * 2 * 2 * 32 * 16 * sizeof(_Float16));
    int*      progress = (int*)     alloc(4 * HH * sizeof(int));

    // --- prep: progress=0, pack weights & input to f16 fragment layouts ---
    init_progress_kernel<<<1, 256, 0, stream>>>(progress);
    pack_weights_kernel<<<(3 * 4 * 10 * 32 * 16) / 256, 256, 0, stream>>>(
        Wx, Ul, Ut, wpack);
    pack_convw_kernel<<<(4 * 2 * 2 * 32 * 16) / 256, 256, 0, stream>>>(
        conv_w, cpack);
    xpack_kernel<<<(int)((cellTotal + 255) / 256), 256, 0, stream>>>(x, xT);

    // --- persistent MDLSTM: one workgroup per (row, direction) ---
    mdlstm_row_kernel<<<dim3(HH, 4), 320, 0, stream>>>(
        xT, wpack, bias, hst, cst, hst16, progress);

    // --- sum 4 directions (unflip) -> f16 ---
    hsum_kernel<<<(int)((cellTotal + 255) / 256), 256, 0, stream>>>(hst, hsum16);

    // --- conv 2x2/s2 + 4*bias + tanh via WMMA GEMM ---
    conv_wmma_kernel<<<dim3((BD * OH * OW) / 16, CO / 16), 32, 0, stream>>>(
        hsum16, cpack, conv_b, y);

    // --- per-channel stats + normalize ---
    stats_kernel<<<CO, 256, 0, stream>>>(y, stats);
    {
        const size_t total = (size_t)BD * CO * OH * OW;
        norm_kernel<<<(int)((total + 255) / 256), 256, 0, stream>>>(
            y, stats, gamma, beta, out);
    }
}
